// SymmetricContraction_45938970198687
// MI455X (gfx1250) — matrix operand order repaired and checked by reading, not compiled, not run on hardware
//
#include <hip/hip_runtime.h>
#include <hip/hip_bf16.h>

typedef __attribute__((ext_vector_type(16))) __bf16 v16bf;
typedef __attribute__((ext_vector_type(8)))  __bf16 v8bf;
typedef __attribute__((ext_vector_type(8)))  float  v8f;

#define NLQ      16
#define CQ       128
#define K3Q      23
#define K2Q      5
#define K1Q      2
#define M_ROWS   480           // 16*23 cubic + 16*5 quad + 16*2 linear rows
#define M_PAD    512           // padded to 32 M-tiles -> 8 per wave, branch-free
#define K_PAD    1024          // 816 cubic + 136 quad + 16 linear + pad
#define NCUBIC   816           // C(18,3)
#define NQUAD    136           // C(17,2)
#define QUAD_OFF 816
#define LIN_OFF  952
#define BASIS_END 968
#define ROW3_END 368           // 16*23
#define ROW2_END 448           // +16*5
#define N_TILES  16384         // (2048*128)/16
#define KCHUNKS  (K_PAD / 32)

// ---------------------------------------------------------------------------
// Symmetrized U value for basis-matrix element (r = (lo,n) row, m = monomial)
// ---------------------------------------------------------------------------
__device__ __forceinline__ float basis_value(
    int r, int m,
    const float* __restrict__ U3, const float* __restrict__ U2,
    const float* __restrict__ U1)
{
    float v = 0.f;
    if (r < ROW3_END && m < NCUBIC) {
        int lo = r / K3Q, n = r % K3Q;
        int mm = m, i = 0, j = 0, k = 0;
        for (i = 0; i < NLQ; i++) { int cnt = (NLQ - i) * (NLQ + 1 - i) / 2; if (mm < cnt) break; mm -= cnt; }
        for (j = i; j < NLQ; j++) { int cnt = NLQ - j; if (mm < cnt) break; mm -= cnt; }
        k = j + mm;
        int per[6][3] = {{i,j,k},{i,k,j},{j,i,k},{j,k,i},{k,i,j},{k,j,i}};
        for (int p = 0; p < 6; p++) {
            bool dup = false;
            for (int q = 0; q < p; q++)
                if (per[q][0]==per[p][0] && per[q][1]==per[p][1] && per[q][2]==per[p][2]) dup = true;
            if (!dup)
                v += U3[(((lo*NLQ + per[p][0])*NLQ + per[p][1])*NLQ + per[p][2])*K3Q + n];
        }
    } else if (r >= ROW3_END && r < ROW2_END && m >= QUAD_OFF && m < QUAD_OFF + NQUAD) {
        int q = r - ROW3_END, lo = q / K2Q, n = q % K2Q;
        int mm = m - QUAD_OFF, i = 0, j = 0;
        for (i = 0; i < NLQ; i++) { int cnt = NLQ - i; if (mm < cnt) break; mm -= cnt; }
        j = i + mm;
        v = U2[((lo*NLQ + i)*NLQ + j)*K2Q + n];
        if (i != j) v += U2[((lo*NLQ + j)*NLQ + i)*K2Q + n];
    } else if (r >= ROW2_END && r < M_ROWS && m >= LIN_OFF && m < LIN_OFF + NLQ) {
        int l = r - ROW2_END, lo = l / K1Q, n = l % K1Q;
        int i = m - LIN_OFF;
        v = U1[(lo*NLQ + i)*K1Q + n];
    }
    return v;   // rows 480..511 and pad columns -> 0
}

// ---------------------------------------------------------------------------
// Kernel 1: build A directly in WMMA A-fragment layout (bf16):
//   A[((mt*KCHUNKS + kc)*32 + lane)*16 + e]
//   lane 0-15  = row M = mt*16+lane,  halfSel=0: e<8 -> K=kc*32+e,      e>=8 -> K=kc*32+16+(e-8)
//   lane 16-31 = row M = mt*16+lane-16, halfSel=1: e<8 -> K=kc*32+8+e,  e>=8 -> K=kc*32+24+(e-8)
//   => a wave's fragment is one contiguous, coalesced 1KB block.
// ---------------------------------------------------------------------------
__global__ __launch_bounds__(256) void build_A_kernel(
    const float* __restrict__ U3, const float* __restrict__ U2,
    const float* __restrict__ U1, __bf16* __restrict__ A)
{
    int idx = blockIdx.x * 256 + threadIdx.x;      // M_PAD/16 * KCHUNKS * 32 * 16 elements
    if (idx >= (M_PAD / 16) * KCHUNKS * 32 * 16) return;
    int e       = idx & 15;
    int lane    = (idx >> 4) & 31;
    int fragIdx = idx >> 9;                        // mt*KCHUNKS + kc
    int kc      = fragIdx % KCHUNKS;
    int mt      = fragIdx / KCHUNKS;
    int halfSel = (lane >> 4) & 1;
    int r = mt * 16 + (lane & 15);
    int m = kc * 32 + halfSel * 8 + (e < 8 ? e : e + 8);
    A[idx] = (__bf16)basis_value(r, m, U3, U2, U1);
}

// ---------------------------------------------------------------------------
// Kernel 2: per 16-wide (b,c) tile:
//   gen monomial matrix Mono[K_PAD][16] (bf16, LDS) -> WMMA GEMM ->
//   F[480][16] (LDS) -> epilogue contraction with W3/W2/W1[atom_type]
// ---------------------------------------------------------------------------
__global__ __launch_bounds__(128) void symcontract_kernel(
    const float* __restrict__ x, const __bf16* __restrict__ A,
    const float* __restrict__ W3, const float* __restrict__ W2,
    const float* __restrict__ W1, const int* __restrict__ atom_types,
    float* __restrict__ out)
{
    __shared__ float  xs[256];                 // 1 KB : x tile [l][col]
    __shared__ __bf16 mono[K_PAD * 16];        // 32 KB: monomial B matrix [K][N]
    __shared__ float  F[M_ROWS * 16];          // 30 KB: GEMM result [M][N]

    const int tid  = threadIdx.x;
    const int tile = blockIdx.x;
    const int b    = tile >> 3;
    const int c0   = (tile & 7) << 4;

    // ---- stage x tile: xs[l*16 + col] = x[b, l, c0+col] ----
    for (int e = tid; e < 256; e += 128)
        xs[e] = x[(b * NLQ + (e >> 4)) * CQ + c0 + (e & 15)];
    __syncthreads();

    // ---- generate symmetric monomial basis (each thread: 1 col, 1/8 of rows) ----
    {
        const int col = tid & 15, slice = tid >> 4;
        int m = 0;
        for (int i = 0; i < NLQ; i++) {
            float xi = xs[i * 16 + col];
            for (int j = i; j < NLQ; j++) {
                float xij = xi * xs[j * 16 + col];
                for (int k = j; k < NLQ; k++) {
                    if ((m & 7) == slice) mono[m * 16 + col] = (__bf16)(xij * xs[k * 16 + col]);
                    m++;
                }
            }
        }
        m = QUAD_OFF;
        for (int i = 0; i < NLQ; i++) {
            float xi = xs[i * 16 + col];
            for (int j = i; j < NLQ; j++) {
                if ((m & 7) == slice) mono[m * 16 + col] = (__bf16)(xi * xs[j * 16 + col]);
                m++;
            }
        }
        for (int i = 0; i < NLQ; i++)
            if (((LIN_OFF + i) & 7) == slice) mono[(LIN_OFF + i) * 16 + col] = (__bf16)xs[i * 16 + col];
        for (int e = tid; e < (K_PAD - BASIS_END) * 16; e += 128)
            mono[(BASIS_END + (e >> 4)) * 16 + (e & 15)] = (__bf16)0.f;
    }
    __syncthreads();

    // ---- WMMA GEMM: F[512][16] = A[512][1024] x Mono[1024][16] ----
    const int lane    = tid & 31;
    const int w       = tid >> 5;              // wave id 0..3; owns M-tiles w, w+4, ..., w+28
    const int rowSel  = lane & 15;
    const int halfSel = (lane >> 4) & 1;

    v8f acc[8];
    #pragma unroll
    for (int t = 0; t < 8; t++) acc[t] = (v8f){0.f,0.f,0.f,0.f,0.f,0.f,0.f,0.f};

    union Frag { v16bf v; v8bf h[2]; };

    auto loadB = [&](int kc) {
        Frag f;
        #pragma unroll
        for (int s = 0; s < 16; s++) {
            int krow = kc * 32 + (halfSel << 3) + (s < 8 ? s : s + 8);
            f.v[s] = mono[krow * 16 + rowSel];
        }
        return f;
    };

    // A fragment base for this lane: contiguous 32B per lane, 1KB per wave-fragment
    const __bf16* Alane = A + lane * 16;

    Frag bf = loadB(0);
    for (int kc = 0; kc < KCHUNKS; kc++) {
        Frag bn = loadB((kc + 1) & (KCHUNKS - 1));   // double-buffered B (wraps harmlessly)
        #pragma unroll
        for (int t = 0; t < 8; t++) {
            const int mt = w + 4 * t;                // all mt < 32 valid (M padded) -> branch-free
            const __bf16* afrag = Alane + (size_t)(mt * KCHUNKS + kc) * 512;
            __builtin_prefetch(afrag + 512, 0, 1);   // next K-chunk fragment of this M-tile
            Frag af;
            af.h[0] = *(const v8bf*)(afrag);
            af.h[1] = *(const v8bf*)(afrag + 8);
            acc[t] = __builtin_amdgcn_wmma_f32_16x16x32_bf16(
                false, af.v, false, bf.v, (short)0, acc[t], false, false);
        }
        bf = bn;
    }

    // ---- spill F (C/D layout: VGPR r -> M = r + halfSel*8, N = lane&15) ----
    #pragma unroll
    for (int t = 0; t < 8; t++) {
        int mt = w + 4 * t;
        if (mt < 30) {                               // tiles 30,31 are padding
            int r0 = mt * 16 + halfSel * 8;
            #pragma unroll
            for (int r = 0; r < 8; r++)
                F[(r0 + r) * 16 + rowSel] = acc[t][r];
        }
    }
    __syncthreads();

    // ---- epilogue: out[b,lo,c] = sum_n F[(lo,n)] * W[lo, type(b), n, c] ----
    const int tb  = atom_types[b];
    const int col = tid & 15;
    const int c   = c0 + col;
    for (int lo = tid >> 4; lo < NLQ; lo += 8) {
        float o = 0.f;
        #pragma unroll
        for (int n = 0; n < K3Q; n++)
            o += F[(lo * K3Q + n) * 16 + col] * W3[((lo * 4 + tb) * K3Q + n) * CQ + c];
        #pragma unroll
        for (int n = 0; n < K2Q; n++)
            o += F[(ROW3_END + lo * K2Q + n) * 16 + col] * W2[((lo * 4 + tb) * K2Q + n) * CQ + c];
        #pragma unroll
        for (int n = 0; n < K1Q; n++)
            o += F[(ROW2_END + lo * K1Q + n) * 16 + col] * W1[((lo * 4 + tb) * K1Q + n) * CQ + c];
        out[(b * NLQ + lo) * CQ + c] = o;
    }
}

extern "C" void kernel_launch(void* const* d_in, const int* in_sizes, int n_in,
                              void* d_out, int out_size, void* d_ws, size_t ws_size,
                              hipStream_t stream) {
    const float* x  = (const float*)d_in[0];
    const float* U3 = (const float*)d_in[1];
    const float* U2 = (const float*)d_in[2];
    const float* U1 = (const float*)d_in[3];
    const float* W3 = (const float*)d_in[4];
    const float* W2 = (const float*)d_in[5];
    const float* W1 = (const float*)d_in[6];
    const int* atom_types = (const int*)d_in[7];
    float* out = (float*)d_out;

    __bf16* A = (__bf16*)d_ws;   // 512*1024*2 = 1,048,576 bytes of scratch

    const int a_elems = (M_PAD / 16) * KCHUNKS * 32 * 16;   // 524288
    build_A_kernel<<<(a_elems + 255) / 256, 256, 0, stream>>>(U3, U2, U1, A);
    symcontract_kernel<<<N_TILES, 128, 0, stream>>>(x, A, W3, W2, W1, atom_types, out);
}
